// RNN_2576980377955
// MI455X (gfx1250) — matrix operand order repaired and checked by reading, not compiled, not run on hardware
//
#include <hip/hip_runtime.h>
#include <stdint.h>

// ---------------------------------------------------------------------------
// CDNA5 / gfx1250 WMMA types
// ---------------------------------------------------------------------------
typedef __attribute__((ext_vector_type(16))) __bf16 v16bf;
typedef __attribute__((ext_vector_type(8)))  float  v8f;
typedef __attribute__((ext_vector_type(4)))  unsigned int u32x4;

union FragU {
    v16bf v;
    u32x4 q[2];
};

// Round-to-nearest-even fp32 -> bf16
static __device__ inline __bf16 f2b(float f) {
    unsigned u = __float_as_uint(f);
    u += 0x7FFFu + ((u >> 16) & 1u);
    unsigned short s = (unsigned short)(u >> 16);
    __bf16 b;
    __builtin_memcpy(&b, &s, 2);
    return b;
}

// A-matrix fragment, 16(M) x 32(K) bf16, source row-major [M][K] at (m0,k0)=base.
// ISA layout: lanes 0-15 (M=lane): v0-3 = K 0..7,  v4-7 = K 16..23
//             lanes 16-31(M=lane-16): v0-3 = K 8..15, v4-7 = K 24..31
static __device__ inline v16bf load_a_frag(const __bf16* base, int ld, int lane) {
    const int half = lane >> 4, r = lane & 15;
    const __bf16* p = base + (size_t)r * ld + half * 8;
    FragU f;
    f.q[0] = *(const u32x4*)(p);        // K half*8 + 0..7
    f.q[1] = *(const u32x4*)(p + 16);   // K half*8 + 16..23
    return f.v;
}

// B-matrix fragment, 32(K) x 16(N) bf16, source row-major [N][K] at (n0,k0)=base.
// ISA layout: lanes 0-15 (N=lane): K 0..7, 16..23 ; lanes 16-31 (N=lane-16): K 8..15, 24..31
static __device__ inline v16bf load_b_frag(const __bf16* base, int ld, int lane) {
    const int half = lane >> 4, r = lane & 15;
    const __bf16* p = base + (size_t)r * ld + half * 8;
    FragU f;
    f.q[0] = *(const u32x4*)(p);
    f.q[1] = *(const u32x4*)(p + 16);
    return f.v;
}

static __device__ inline v8f wmma_bf16(v16bf a, v16bf b, v8f c) {
    return __builtin_amdgcn_wmma_f32_16x16x32_bf16(
        /*neg_a=*/false, a, /*neg_b=*/false, b,
        /*c_mod=*/(short)0, c, /*reuse_a=*/false, /*reuse_b=*/false);
}

// ---------------------------------------------------------------------------
// Constants
// ---------------------------------------------------------------------------
#define VV 32000
#define EE 512
#define HH 512
#define OO 4
#define BB 64
#define TT 256

// ---------------------------------------------------------------------------
// Kernel 1: embedding gather + cast to bf16.  eb[b*T+t][k] = bf16(emb[x[b][t]][k])
// ---------------------------------------------------------------------------
__global__ __launch_bounds__(256) void embed_cast_kernel(
    const int* __restrict__ x, const float* __restrict__ emb,
    __bf16* __restrict__ eb)
{
    const int r = blockIdx.x;                 // 0 .. B*T-1
    const int idx = x[r];
    const float* src = emb + (size_t)idx * EE;
    __bf16* dst = eb + (size_t)r * EE;
    for (int k = threadIdx.x; k < EE; k += 256)
        dst[k] = f2b(src[k]);
}

// ---------------------------------------------------------------------------
// Kernel 2: cast W_ih and W_hh to bf16 (layouts preserved: [N][K] row-major)
// ---------------------------------------------------------------------------
__global__ __launch_bounds__(256) void cast_w_kernel(
    const float* __restrict__ wih, const float* __restrict__ whh,
    __bf16* __restrict__ wihb, __bf16* __restrict__ whhb)
{
    const int i = blockIdx.x * 256 + threadIdx.x;   // exactly H*E threads
    wihb[i] = f2b(wih[i]);
    whhb[i] = f2b(whh[i]);
}

// ---------------------------------------------------------------------------
// Kernel 3: xw = e @ W_ih^T + b_ih, written time-major fp32 [T][B][H].
// One 16x16 output tile per wave; K=512 -> 16 WMMA per tile.
// Grid: 4096 blocks x 256 threads (8 waves) = 32768 tiles = (16384/16)*(512/16).
// ---------------------------------------------------------------------------
__global__ __launch_bounds__(256) void gemm_xw_kernel(
    const __bf16* __restrict__ eb,    // [B*T][E]
    const __bf16* __restrict__ wihb,  // [H][E]
    const float*  __restrict__ bih,   // [H]
    float* __restrict__ xw)           // [T][B][H]
{
    const int lane = threadIdx.x & 31;
    const int wave = threadIdx.x >> 5;
    const int tile = blockIdx.x * 8 + wave;
    const int mt = tile >> 5;          // 0..1023
    const int nt = tile & 31;          // 0..31
    const int m0 = mt * 16, n0 = nt * 16;

    v8f c = {};
#pragma unroll
    for (int kt = 0; kt < 16; ++kt) {
        v16bf a = load_a_frag(eb   + (size_t)m0 * EE + kt * 32, EE, lane);
        v16bf b = load_b_frag(wihb + (size_t)n0 * EE + kt * 32, EE, lane);
        c = wmma_bf16(a, b, c);
    }

    const int half = lane >> 4;
    const int col  = n0 + (lane & 15);
    const float bias = bih[col];
#pragma unroll
    for (int r = 0; r < 8; ++r) {
        const int row = m0 + r + half * 8;   // global row = b*T + t
        const int b_ = row >> 8;             // / T (T=256)
        const int t_ = row & 255;            // % T
        xw[((size_t)t_ * BB + b_) * HH + col] = c[r] + bias;
    }
}

// ---------------------------------------------------------------------------
// Kernel 4: recurrent scan. 4 blocks x 512 threads (16 waves).
// Block handles 16 batch rows (one M-tile) for ALL timesteps (batch rows are
// independent -> no inter-block sync). Each wave owns 2 N-tiles (32 h columns).
//
// VGPR budget (128 KB / SIMD32 == 1024 wave32-VGPRs, 4 waves/SIMD):
//   - W_hh B-fragments for kt=0..7 register-resident: 2*8 frags = 128 VGPRs,
//     loaded once before the t-loop, reused 256x.
//   - kt=8..15 streamed from L2 each step (W_hh is 512 KB, L2-resident);
//     the L2 stream overlaps the LDS A-fragment path and the WMMA pipe.
//   ~220 VGPRs/wave total -> 880/SIMD, fits.
//
// h ping-pongs between two LDS buffers (bf16, padded stride) ; two barriers
// per step. Per-step xw values are prefetched into registers at step start so
// their L2 latency hides under the 32 WMMAs. On the final step we skip the
// bf16 h write and store fp32 hT aliased over the (dead) ping-pong buffers.
// ---------------------------------------------------------------------------
__global__ __launch_bounds__(512) void rnn_scan_kernel(
    const __bf16* __restrict__ whhb,  // [H][H] row-major [n][k]
    const float*  __restrict__ xw,    // [T][B][H]
    const float*  __restrict__ bhh,   // [H]
    const float*  __restrict__ wfc,   // [O][H]
    const float*  __restrict__ bfc,   // [O]
    float* __restrict__ out)          // [B][O]
{
    constexpr int LDH = HH + 8;            // bf16 row stride (bank padding)
    constexpr int LDF = HH + 4;            // fp32 row stride for hT
    constexpr int HELEM = 16 * LDH;        // elements per ping buffer
    __shared__ __align__(16) char smem[2 * HELEM * 2];   // 33280 bytes
    __bf16* hbuf = (__bf16*)smem;          // two ping-pong bf16 buffers
    float*  hf   = (float*)smem;           // fp32 hT, aliases smem after final step

    const int lane = threadIdx.x & 31;
    const int wave = threadIdx.x >> 5;     // 0..15
    const int half = lane >> 4;
    const int r15  = lane & 15;
    const int mrow0 = blockIdx.x * 16;     // batch rows [mrow0, mrow0+16)
    const int n0a = (wave * 2 + 0) * 16;
    const int n0b = (wave * 2 + 1) * 16;

    // Register-resident half of this wave's W_hh slice (kt = 0..7).
    v16bf bw0[8], bw1[8];
#pragma unroll
    for (int kt = 0; kt < 8; ++kt) {
        bw0[kt] = load_b_frag(whhb + (size_t)n0a * HH + kt * 32, HH, lane);
        bw1[kt] = load_b_frag(whhb + (size_t)n0b * HH + kt * 32, HH, lane);
    }
    const float bias0 = bhh[n0a + r15];
    const float bias1 = bhh[n0b + r15];

    // h0 = 0 (buffer 0)
    for (int i = threadIdx.x; i < HELEM; i += 512)
        hbuf[i] = f2b(0.0f);
    __syncthreads();

    for (int t = 0; t < TT; ++t) {
        const __bf16* hr = hbuf + (t & 1) * HELEM;        // read h_{t}
        __bf16*       hw = hbuf + ((t + 1) & 1) * HELEM;  // write h_{t+1}

        // Prefetch this step's xw values (hides L2 latency under WMMAs).
        float xv0[8], xv1[8];
#pragma unroll
        for (int rr = 0; rr < 8; ++rr) {
            const size_t base = ((size_t)t * BB + mrow0 + rr + half * 8) * HH;
            xv0[rr] = xw[base + n0a + r15];
            xv1[rr] = xw[base + n0b + r15];
        }

        v8f c0 = {}, c1 = {};
#pragma unroll
        for (int kt = 0; kt < 8; ++kt) {                  // register B half
            v16bf a = load_a_frag(hr + kt * 32, LDH, lane);
            c0 = wmma_bf16(a, bw0[kt], c0);
            c1 = wmma_bf16(a, bw1[kt], c1);
        }
#pragma unroll
        for (int kt = 8; kt < 16; ++kt) {                 // L2-streamed B half
            v16bf b0 = load_b_frag(whhb + (size_t)n0a * HH + kt * 32, HH, lane);
            v16bf b1 = load_b_frag(whhb + (size_t)n0b * HH + kt * 32, HH, lane);
            v16bf a  = load_a_frag(hr + kt * 32, LDH, lane);
            c0 = wmma_bf16(a, b0, c0);
            c1 = wmma_bf16(a, b1, c1);
        }
        __syncthreads();   // all reads of hr done; write buffer is safe

        if (t < TT - 1) {
#pragma unroll
            for (int rr = 0; rr < 8; ++rr) {
                const int row = rr + half * 8;
                hw[row * LDH + n0a + r15] = f2b(tanhf(c0[rr] + xv0[rr] + bias0));
                hw[row * LDH + n0b + r15] = f2b(tanhf(c1[rr] + xv1[rr] + bias1));
            }
        } else {
            // Final step: both ping buffers are dead; store fp32 hT over them.
#pragma unroll
            for (int rr = 0; rr < 8; ++rr) {
                const int row = rr + half * 8;
                hf[row * LDF + n0a + r15] = tanhf(c0[rr] + xv0[rr] + bias0);
                hf[row * LDF + n0b + r15] = tanhf(c1[rr] + xv1[rr] + bias1);
            }
        }
        __syncthreads();   // h_{t+1} (or hT) fully published
    }

    // Final FC: out = hT @ W_fc^T + b_fc.
    // 64 tasks (16 rows x 4 outputs), 8 threads per task, shuffle reduction.
    {
        const int task = threadIdx.x >> 3;   // 0..63
        const int seg  = threadIdx.x & 7;    // 0..7
        const int row  = task >> 2;          // 0..15
        const int o    = task & 3;           // 0..3
        float s = 0.0f;
        const int k0 = seg * 64;
#pragma unroll 8
        for (int k = k0; k < k0 + 64; ++k)
            s += hf[row * LDF + k] * wfc[o * HH + k];
        s += __shfl_xor(s, 4, 32);
        s += __shfl_xor(s, 2, 32);
        s += __shfl_xor(s, 1, 32);
        if (seg == 0)
            out[(mrow0 + row) * OO + o] = s + bfc[o];
    }
}

// ---------------------------------------------------------------------------
// Host launcher
// ---------------------------------------------------------------------------
extern "C" void kernel_launch(void* const* d_in, const int* in_sizes, int n_in,
                              void* d_out, int out_size, void* d_ws, size_t ws_size,
                              hipStream_t stream) {
    (void)in_sizes; (void)n_in; (void)out_size; (void)ws_size;

    const int*   x    = (const int*)d_in[0];
    const float* emb  = (const float*)d_in[1];
    const float* wih  = (const float*)d_in[2];
    const float* whh  = (const float*)d_in[3];
    const float* bih  = (const float*)d_in[4];
    const float* bhh  = (const float*)d_in[5];
    const float* wfc  = (const float*)d_in[6];
    const float* bfc  = (const float*)d_in[7];
    float* out = (float*)d_out;

    char* ws = (char*)d_ws;
    // Workspace layout (49 MB total):
    //   [0, 16MB)        e_bf16   [B*T][E]
    //   [16MB, 16.5MB)   W_ih bf16 [H][E]
    //   [16.5MB, 17MB)   W_hh bf16 [H][H]
    //   [17MB, 49MB)     xw fp32  [T][B][H]
    __bf16* eb   = (__bf16*)(ws);
    __bf16* wihb = (__bf16*)(ws + ((size_t)16 << 20));
    __bf16* whhb = (__bf16*)(ws + ((size_t)16 << 20) + ((size_t)512 << 10));
    float*  xw   = (float*) (ws + ((size_t)17 << 20));

    embed_cast_kernel<<<BB * TT, 256, 0, stream>>>(x, emb, eb);
    cast_w_kernel<<<(HH * EE) / 256, 256, 0, stream>>>(wih, whh, wihb, whhb);
    gemm_xw_kernel<<<(BB * TT / 16) * (HH / 16) / 8, 256, 0, stream>>>(eb, wihb, bih, xw);
    rnn_scan_kernel<<<BB / 16, 512, 0, stream>>>(whhb, xw, bhh, wfc, bfc, out);
}
